// InterpretableMultiHeadAttention_41042707480985
// MI455X (gfx1250) — matrix operand order repaired and checked by reading, not compiled, not run on hardware
//
#include <hip/hip_runtime.h>
#include <hip/hip_bf16.h>

// Sizes (fixed by the reference)
#define B_  2
#define S_  2048
#define D_  512
#define H_  8
#define DH_ 64

typedef _Float16 f16;
typedef _Float16 v16h __attribute__((ext_vector_type(16)));
typedef _Float16 v8h  __attribute__((ext_vector_type(8)));
typedef float    v8f  __attribute__((ext_vector_type(8)));
typedef float    v4f  __attribute__((ext_vector_type(4)));

union V16H { v16h v; v8h h[2]; };

#define LT_STRIDE 2052   // f32 logits rows: 2048 + 4 pad (bank-conflict-free)
#define PH_STRIDE 2056   // f16 probs rows: 2048 + 8 pad (16B-aligned chunks, 4-bank row skew)

__device__ __forceinline__ float wave_sum(float x) {
#pragma unroll
  for (int off = 16; off > 0; off >>= 1) x += __shfl_xor(x, off, 32);
  return x;
}

// ---------------------------------------------------------------------------
// Kernel 0: fp32 -> f16 conversions; weights stored transposed ([n][k]) so
// WMMA B-fragments are contiguous 32B loads per lane.
// ---------------------------------------------------------------------------
__global__ __launch_bounds__(256) void k_convert(
    const float* __restrict__ x, const float* __restrict__ Wq,
    const float* __restrict__ Wk, const float* __restrict__ Wv,
    const float* __restrict__ Wout,
    f16* __restrict__ xh, f16* __restrict__ wqt, f16* __restrict__ wkt,
    f16* __restrict__ wvt, f16* __restrict__ wot) {
  long i = (long)blockIdx.x * 256 + threadIdx.x;
  if (i < 2097152) { xh[i] = (f16)x[i]; return; }
  i -= 2097152;
  if (i < 262144) { long k = i >> 9, n = i & 511; wqt[n * 512 + k] = (f16)Wq[i]; return; }
  i -= 262144;
  if (i < 262144) { long k = i >> 9, n = i & 511; wkt[n * 512 + k] = (f16)Wk[i]; return; }
  i -= 262144;
  if (i < 32768)  { long k = i >> 6, n = i & 63;  wvt[n * 512 + k] = (f16)Wv[i]; return; }
  i -= 32768;
  if (i < 262144) { long k = i >> 9, n = i & 511; wot[n * 512 + k] = (f16)Wout[i]; return; }
}

// ---------------------------------------------------------------------------
// Kernel 1: fused QKV projection. One wave = one 16x16 C tile.
// Columns 0..511 -> Q ([B,H,S,DH] f16), 512..1023 -> K, 1024..1087 -> V
// (written transposed as Vt[b][dh][s] for the P@V B-fragments).
// ---------------------------------------------------------------------------
__global__ __launch_bounds__(256) void k_qkv(
    const f16* __restrict__ xh, const f16* __restrict__ wqt,
    const f16* __restrict__ wkt, const f16* __restrict__ wvt,
    const float* __restrict__ bq, const float* __restrict__ bk,
    const float* __restrict__ bv,
    f16* __restrict__ Qh, f16* __restrict__ Kh, f16* __restrict__ Vt) {
  int wave = threadIdx.x >> 5, lane = threadIdx.x & 31;
  int id = blockIdx.x * 8 + wave;          // 0..17407 = 256 mtiles * 68 ntiles
  int mt = id / 68, nt = id % 68;
  int m0 = mt * 16, n0 = nt * 16;
  int rc = lane & 15, kh = lane >> 4;

  const f16* BtBase; const float* bias; int nloc;
  if (n0 < 512)       { BtBase = wqt; bias = bq; nloc = n0; }
  else if (n0 < 1024) { BtBase = wkt; bias = bk; nloc = n0 - 512; }
  else                { BtBase = wvt; bias = bv; nloc = n0 - 1024; }

  const f16* Arow = xh     + (long)(m0 + rc)   * 512 + kh * 16;
  const f16* Brow = BtBase + (long)(nloc + rc) * 512 + kh * 16;

  v8f acc = {};
#pragma unroll 4
  for (int kb = 0; kb < 512; kb += 32) {
    V16H a, bfr;
    a.h[0] = *(const v8h*)(Arow + kb);
    a.h[1] = *(const v8h*)(Arow + kb + 8);
    bfr.v  = *(const v16h*)(Brow + kb);
    acc = __builtin_amdgcn_wmma_f32_16x16x32_f16(false, a.v, false, bfr.v,
                                                 (short)0, acc, false, false);
  }
  float bval = bias[nloc + rc];
  int nc = nloc + rc;
  int mBase = m0 + kh * 8;
#pragma unroll
  for (int r = 0; r < 8; r++) {
    int m = mBase + r;
    int bi = m >> 11, sl = m & 2047;
    float vv = acc[r] + bval;
    if (n0 < 512)
      Qh[((long)(bi * 8 + (nc >> 6)) * 2048 + sl) * 64 + (nc & 63)] = (f16)vv;
    else if (n0 < 1024)
      Kh[((long)(bi * 8 + (nc >> 6)) * 2048 + sl) * 64 + (nc & 63)] = (f16)vv;
    else
      Vt[(long)(bi * 64 + nc) * 2048 + sl] = (f16)vv;
  }
}

// ---------------------------------------------------------------------------
// Kernel 2: fused attention. One block = one (b, 16-query tile); loops heads.
//   phase 1: logits (WMMA QK^T) -> LDS f32 (16 x 2048) + K-stream prefetch
//   phase 2: sparsemax via Michelot iteration with exact early exit;
//            writeback emits both f32 (for avg) and f16 (for WMMA) copies
//   phase 3/4 (overlapped): waves 4..7 do avg_attention RMW (L2-resident)
//            while waves 0..3 do heads = P @ V with pure ds_load A-fragments
// ---------------------------------------------------------------------------
__global__ __launch_bounds__(256) void k_attn(
    const f16* __restrict__ Qh, const f16* __restrict__ Kh,
    const f16* __restrict__ Vt, f16* __restrict__ headsh,
    float* __restrict__ avg_out) {
  __shared__ float lt[16 * LT_STRIDE];   // 128.25 KB: f32 logits/probs
  __shared__ f16   ph[16 * PH_STRIDE];   //  64.25 KB: f16 probs (A-fragments)

  int b  = blockIdx.x >> 7;
  int qt = blockIdx.x & 127;
  int q0 = qt * 16;
  int wave = threadIdx.x >> 5, lane = threadIdx.x & 31;
  int rc = lane & 15, kh = lane >> 4;

  for (int h = 0; h < H_; ++h) {
    const f16* Qbase = Qh + ((long)(b * H_ + h) * S_ + q0) * DH_;
    const f16* Kbase = Kh + ((long)(b * H_ + h) * S_) * DH_;

    // A-fragments (16 query rows, d=0..31 and d=32..63), hoisted per head
    V16H a0, a1;
    {
      const f16* qr = Qbase + rc * DH_;
      a0.h[0] = *(const v8h*)(qr + kh * 16);
      a0.h[1] = *(const v8h*)(qr + 8 + kh * 16);
      a1.h[0] = *(const v8h*)(qr + 32 + kh * 16);
      a1.h[1] = *(const v8h*)(qr + 40 + kh * 16);
    }

    // ---- phase 1: logits -> LDS -------------------------------------------
    for (int t = wave; t < 128; t += 8) {
      int k0 = t * 16;
      const f16* krow = Kbase + (long)(k0 + rc) * DH_ + kh * 16;
      if (t + 8 < 128)  // prefetch this wave's next K tile (global_prefetch_b8)
        __builtin_prefetch(krow + 128 * DH_, 0, 1);
      V16H b0, b1;
      b0.v = *(const v16h*)(krow);
      b1.v = *(const v16h*)(krow + 32);
      v8f acc = {};
      acc = __builtin_amdgcn_wmma_f32_16x16x32_f16(false, a0.v, false, b0.v,
                                                   (short)0, acc, false, false);
      acc = __builtin_amdgcn_wmma_f32_16x16x32_f16(false, a1.v, false, b1.v,
                                                   (short)0, acc, false, false);
      int mb = kh * 8;
#pragma unroll
      for (int r = 0; r < 8; r++) lt[(mb + r) * LT_STRIDE + k0 + rc] = acc[r];
    }
    __syncthreads();

    // ---- phase 2: sparsemax (Michelot + exact early exit), 2 rows/wave ----
#pragma unroll
    for (int rr = 0; rr < 2; rr++) {
      int row = wave * 2 + rr;
      float zc[64];
      float s0 = 0.f;
#pragma unroll
      for (int j = 0; j < 64; j++) {
        zc[j] = lt[row * LT_STRIDE + j * 32 + lane];
        s0 += zc[j];
      }
      float tau = (wave_sum(s0) - 1.0f) * (1.0f / 2048.0f);
      float cprev = 2049.0f;                 // impossible count -> enter loop
      for (int it = 0; it < 48; ++it) {
        float s = 0.f, c = 0.f;
#pragma unroll
        for (int j = 0; j < 64; j++) {
          float z = zc[j];
          if (z > tau) { s += z; c += 1.0f; }
        }
        s = wave_sum(s); c = wave_sum(c);    // bitwise-uniform across lanes
        tau = (s - 1.0f) / c;                // c >= 1 always (tau < max(z))
        if (c == cprev) break;               // active set stable => converged
        cprev = c;
      }
#pragma unroll
      for (int j = 0; j < 64; j++) {
        float p = fmaxf(zc[j] - tau, 0.0f);
        lt[row * LT_STRIDE + j * 32 + lane] = p;       // f32 copy (avg output)
        ph[row * PH_STRIDE + j * 32 + lane] = (f16)p;  // f16 copy (WMMA A)
      }
    }
    __syncthreads();

    // ---- phases 3+4, overlapped across wave halves ------------------------
    if (wave >= 4) {
      // avg_attention accumulation: per-thread RMW mapping is identical every
      // head, so h>0 reads this thread's own prior writes (L2-resident).
      float* avgb = avg_out + (long)(b * S_ + q0) * S_;
      int t = threadIdx.x - 128;
      for (int i = t; i < (16 * 2048 / 4); i += 128) {
        int row = i >> 9;
        int col = (i & 511) << 2;
        v4f p = *(const v4f*)(lt + row * LT_STRIDE + col);
        float* dp = avgb + (long)row * S_ + col;
        v4f o;
        if (h == 0) {
          o = p * 0.125f;
        } else {
          v4f prev = *(const v4f*)dp;
          o = prev + p * 0.125f;
        }
        *(v4f*)dp = o;
      }
    } else {
      // heads = P @ V : wave w -> output columns [w*16, w*16+16)
      int n0 = wave * 16;
      const f16* vrow = Vt + ((long)(b * 64) + n0 + rc) * 2048 + kh * 16;
      const f16* prow = ph + rc * PH_STRIDE + kh * 16;
      v8f acc = {};
      for (int ks = 0; ks < 64; ++ks) {
        int kb = ks * 32;
        V16H a, bb;
        a.h[0] = *(const v8h*)(prow + kb);
        a.h[1] = *(const v8h*)(prow + kb + 8);
        bb.v   = *(const v16h*)(vrow + kb);
        acc = __builtin_amdgcn_wmma_f32_16x16x32_f16(false, a.v, false, bb.v,
                                                     (short)0, acc, false, false);
      }
      int mb = kh * 8;
#pragma unroll
      for (int r = 0; r < 8; r++)
        headsh[((long)(b * S_) + q0 + mb + r) * D_ + h * DH_ + n0 + rc] = (f16)acc[r];
    }
    __syncthreads();   // protect lt/ph before next head overwrites them
  }
}

// ---------------------------------------------------------------------------
// Kernel 3: x_out = heads @ Wout + bout (f16 WMMA, f32 out)
// ---------------------------------------------------------------------------
__global__ __launch_bounds__(256) void k_out(
    const f16* __restrict__ headsh, const f16* __restrict__ wot,
    const float* __restrict__ bout, float* __restrict__ xout) {
  int wave = threadIdx.x >> 5, lane = threadIdx.x & 31;
  int id = blockIdx.x * 8 + wave;          // 0..8191 = 256 mtiles * 32 ntiles
  int mt = id >> 5, nt = id & 31;
  int m0 = mt * 16, n0 = nt * 16;
  int rc = lane & 15, kh = lane >> 4;

  const f16* Arow = headsh + (long)(m0 + rc) * 512 + kh * 16;
  const f16* Brow = wot    + (long)(n0 + rc) * 512 + kh * 16;

  v8f acc = {};
#pragma unroll 4
  for (int kb = 0; kb < 512; kb += 32) {
    V16H a, bfr;
    a.h[0] = *(const v8h*)(Arow + kb);
    a.h[1] = *(const v8h*)(Arow + kb + 8);
    bfr.v  = *(const v16h*)(Brow + kb);
    acc = __builtin_amdgcn_wmma_f32_16x16x32_f16(false, a.v, false, bfr.v,
                                                 (short)0, acc, false, false);
  }
  float bv = bout[n0 + rc];
  int mb = m0 + kh * 8;
#pragma unroll
  for (int r = 0; r < 8; r++)
    xout[(long)(mb + r) * 512 + n0 + rc] = acc[r] + bv;
}

// ---------------------------------------------------------------------------
extern "C" void kernel_launch(void* const* d_in, const int* in_sizes, int n_in,
                              void* d_out, int out_size, void* d_ws, size_t ws_size,
                              hipStream_t stream) {
  (void)in_sizes; (void)n_in; (void)out_size; (void)ws_size;
  const float* x    = (const float*)d_in[0];
  const float* Wq   = (const float*)d_in[1];
  const float* bq   = (const float*)d_in[2];
  const float* Wk   = (const float*)d_in[3];
  const float* bk   = (const float*)d_in[4];
  const float* Wv   = (const float*)d_in[5];
  const float* bv   = (const float*)d_in[6];
  const float* Wout = (const float*)d_in[7];
  const float* bout = (const float*)d_in[8];
  float* out = (float*)d_out;

  char* ws = (char*)d_ws;                 // ~18.1 MB total scratch
  f16* xh  = (f16*)(ws);                  // 4,194,304 B
  f16* wqt = (f16*)(ws + 4194304);        //   524,288 B
  f16* wkt = (f16*)(ws + 4718592);        //   524,288 B
  f16* wvt = (f16*)(ws + 5242880);        //    65,536 B
  f16* wot = (f16*)(ws + 5308416);        //   524,288 B
  f16* Qh  = (f16*)(ws + 5832704);        // 4,194,304 B
  f16* Kh  = (f16*)(ws + 10027008);       // 4,194,304 B
  f16* Vt  = (f16*)(ws + 14221312);       //   524,288 B
  f16* hh  = (f16*)(ws + 14745600);       // 4,194,304 B

  k_convert<<<11392, 256, 0, stream>>>(x, Wq, Wk, Wv, Wout, xh, wqt, wkt, wvt, wot);
  k_qkv<<<2176, 256, 0, stream>>>(xh, wqt, wkt, wvt, bq, bk, bv, Qh, Kh, Vt);
  k_attn<<<256, 256, 0, stream>>>(Qh, Kh, Vt, hh, out + 2097152);
  k_out<<<1024, 256, 0, stream>>>(hh, wot, bout, out);
}
